// ImageTextModule_37735582662685
// MI455X (gfx1250) — compile-verified, matrix operand order checked by reading, not run on hardware
//
#include <hip/hip_runtime.h>
#include <math.h>

// ---------------------------------------------------------------------------
// CDNA5 / gfx1250 image-text cross attention pipeline (bf16 WMMA, f32 accum)
// Double-buffered LDS, async global->LDS copies on the pure-copy path.
// ---------------------------------------------------------------------------

typedef __attribute__((ext_vector_type(16))) __bf16 v16bf;
typedef __attribute__((ext_vector_type(8)))  __bf16 v8bf;
typedef __attribute__((ext_vector_type(8)))  float  v8f;

#define NT_  4096
#define NI_  4096
#define HID_ 1024
#define DIM_ 1024

#define BM 128
#define BN 128
#define BK 32
#define LDT 40   // LDS row stride in bf16 elems: 32 + 8 pad (80B, keeps 16B align)

// Fill one 16x32 bf16 fragment from an LDS tile laid out row-major (stride LDT),
// matching the ISA 16-bit A-matrix layout (CDNA5 ISA 7.12.2):
//   lane<16  : row=lane,    K 0..7  in v0..3, K 16..23 in v4..7
//   lane>=16 : row=lane-16, K 8..15 in v0..3, K 24..31 in v4..7
// -> per lane: two contiguous 16B chunks => two ds_load_b128.
__device__ __forceinline__ v16bf frag_ld(const __bf16* tile, int lane) {
  const int h = lane >> 4;
  const int r = lane & 15;
  const __bf16* row = tile + r * LDT;
  v8bf lo = *(const v8bf*)(row + 8 * h);
  v8bf hi = *(const v8bf*)(row + 16 + 8 * h);
  v16bf f;
#pragma unroll
  for (int j = 0; j < 8; ++j) { f[j] = lo[j]; f[8 + j] = hi[j]; }
  return f;
}

__device__ __forceinline__ v8f wmma_bf16(v16bf a, v16bf b, v8f c) {
  // (neg_a, A, neg_b, B, c_mod, C, reuse_a, reuse_b)
  return __builtin_amdgcn_wmma_f32_16x16x32_bf16(false, a, false, b, (short)0, c,
                                                 false, false);
}

// One 32x64-per-wave MMA step: 6 fragment loads -> 8 WMMAs.
__device__ __forceinline__ void mma_step(const __bf16* sA, const __bf16* sB,
                                         int wm, int wn, int lane,
                                         v8f acc[2][4]) {
  v16bf a0 = frag_ld(sA + wm * LDT, lane);
  v16bf a1 = frag_ld(sA + (wm + 16) * LDT, lane);
#pragma unroll
  for (int j = 0; j < 4; ++j) {
    v16bf b = frag_ld(sB + (wn + 16 * j) * LDT, lane);
    acc[0][j] = wmma_bf16(a0, b, acc[0][j]);
    acc[1][j] = wmma_bf16(a1, b, acc[1][j]);
  }
}

// CDNA5 async copy: 16B global -> LDS, tracked on ASYNCcnt (CDNA5 ISA).
// The LDS operand takes the wave-relative LDS byte address == low 32 bits of the
// generic pointer (aperture mapping).
__device__ __forceinline__ void async_cp_b128(void* lds_dst, const void* gsrc) {
  unsigned lds = (unsigned)(unsigned long long)lds_dst;
  asm volatile("global_load_async_to_lds_b128 %0, %1, off"
               :
               : "v"(lds), "v"(gsrc)
               : "memory");
}
__device__ __forceinline__ void wait_async0() {
  asm volatile("s_wait_asynccnt 0x0" ::: "memory");
}

// ---------------------------------------------------------------------------
// f32 -> bf16 elementwise convert (for raw feature matrices used as GEMM V)
// ---------------------------------------------------------------------------
__global__ void cvt_bf16_kernel(const float* __restrict__ x,
                                __bf16* __restrict__ y, long n) {
  long i = (long)blockIdx.x * blockDim.x + threadIdx.x;
  long stride = (long)gridDim.x * blockDim.x;
  for (; i < n; i += stride) y[i] = (__bf16)x[i];
}

// ---------------------------------------------------------------------------
// Projection: Y_bf16[M,N] = X[M,K] @ W[N,K]^T + bias[N]   (f32 in, bf16 out)
// VGPR staging (needs f32->bf16 convert), double-buffered LDS.
// ---------------------------------------------------------------------------
__global__ __launch_bounds__(256, 2) void proj_kernel(
    const float* __restrict__ X, const float* __restrict__ W,
    const float* __restrict__ bias, __bf16* __restrict__ Y,
    int M, int N, int K) {
  __shared__ __bf16 sA[2][BM * LDT];
  __shared__ __bf16 sB[2][BN * LDT];
  const int tid = threadIdx.x, lane = tid & 31, wid = tid >> 5;
  const int wm = (wid & 3) * 32, wn = (wid >> 2) * 64;
  const long bm0 = (long)blockIdx.y * BM, bn0 = (long)blockIdx.x * BN;

  v8f acc[2][4];
#pragma unroll
  for (int i = 0; i < 2; ++i)
#pragma unroll
    for (int j = 0; j < 4; ++j)
#pragma unroll
      for (int v = 0; v < 8; ++v) acc[i][j][v] = 0.0f;

  const int sr = tid >> 1;          // 0..127
  const int sc = (tid & 1) * 16;    // 0 or 16

  auto stage = [&](int buf, int k0) {
    const float* srcA = X + (bm0 + sr) * (long)K + k0 + sc;
    __bf16* dstA = sA[buf] + sr * LDT + sc;
#pragma unroll
    for (int j = 0; j < 16; ++j) dstA[j] = (__bf16)srcA[j];
    const float* srcB = W + (bn0 + sr) * (long)K + k0 + sc;
    __bf16* dstB = sB[buf] + sr * LDT + sc;
#pragma unroll
    for (int j = 0; j < 16; ++j) dstB[j] = (__bf16)srcB[j];
  };

  stage(0, 0);
  __syncthreads();
  for (int k0 = 0; k0 < K; k0 += BK) {
    const int cur = (k0 / BK) & 1;
    if (k0 + BK < K) stage(cur ^ 1, k0 + BK);   // overlap with WMMA below
    mma_step(sA[cur], sB[cur], wm, wn, lane, acc);
    __syncthreads();                             // one barrier per K-step
  }

  // Epilogue per ISA C/D layout: lane<16 holds M=v, lane>=16 holds M=v+8.
  const int nl = lane & 15, h = lane >> 4;
#pragma unroll
  for (int j = 0; j < 4; ++j) {
    const long col = bn0 + wn + 16 * j + nl;
    const float bv = bias[col];
#pragma unroll
    for (int i = 0; i < 2; ++i)
#pragma unroll
      for (int v = 0; v < 8; ++v) {
        const long row = bm0 + wm + 16 * i + 8 * h + v;
        Y[row * (long)N + col] = (__bf16)(acc[i][j][v] + bv);
      }
  }
}

// ---------------------------------------------------------------------------
// Logits: L[M,N] = (T[M,K] @ I[N,K]^T) * (1/sqrt(H))   (bf16 in, f32 out)
// Pure-copy staging -> CDNA5 async global->LDS (no VGPR round trip),
// double-buffered, s_wait_asynccnt + one barrier per K-step.
// ---------------------------------------------------------------------------
__global__ __launch_bounds__(256, 2) void logits_kernel(
    const __bf16* __restrict__ Tm, const __bf16* __restrict__ Im,
    float* __restrict__ L, int M, int N, int K) {
  __shared__ __bf16 sA[2][BM * LDT];
  __shared__ __bf16 sB[2][BN * LDT];
  const int tid = threadIdx.x, lane = tid & 31, wid = tid >> 5;
  const int wm = (wid & 3) * 32, wn = (wid >> 2) * 64;
  const long bm0 = (long)blockIdx.y * BM, bn0 = (long)blockIdx.x * BN;

  v8f acc[2][4];
#pragma unroll
  for (int i = 0; i < 2; ++i)
#pragma unroll
    for (int j = 0; j < 4; ++j)
#pragma unroll
      for (int v = 0; v < 8; ++v) acc[i][j][v] = 0.0f;

  const int sr = tid >> 1;
  const int sc = (tid & 1) * 16;

  auto stage = [&](int buf, int k0) {
    const __bf16* srcA = Tm + (bm0 + sr) * (long)K + k0 + sc;
    async_cp_b128(sA[buf] + sr * LDT + sc, srcA);
    async_cp_b128(sA[buf] + sr * LDT + sc + 8, srcA + 8);
    const __bf16* srcB = Im + (bn0 + sr) * (long)K + k0 + sc;
    async_cp_b128(sB[buf] + sr * LDT + sc, srcB);
    async_cp_b128(sB[buf] + sr * LDT + sc + 8, srcB + 8);
  };

  stage(0, 0);
  wait_async0();
  __syncthreads();
  for (int k0 = 0; k0 < K; k0 += BK) {
    const int cur = (k0 / BK) & 1;
    if (k0 + BK < K) stage(cur ^ 1, k0 + BK);   // async copies run under WMMAs
    mma_step(sA[cur], sB[cur], wm, wn, lane, acc);
    wait_async0();                               // own copies landed in LDS
    __syncthreads();                             // publish across waves
  }

  const int nl = lane & 15, h = lane >> 4;
  const float inv_sqrt_h = 0.03125f;  // 1/sqrt(1024)
#pragma unroll
  for (int j = 0; j < 4; ++j) {
    const long col = bn0 + wn + 16 * j + nl;
#pragma unroll
    for (int i = 0; i < 2; ++i)
#pragma unroll
      for (int v = 0; v < 8; ++v) {
        const long row = bm0 + wm + 16 * i + 8 * h + v;
        L[row * (long)N + col] = acc[i][j][v] * inv_sqrt_h;
      }
  }
}

// ---------------------------------------------------------------------------
// Softmax stats over rows (max + 1/sum) — one block per row, coalesced reads
// ---------------------------------------------------------------------------
__global__ __launch_bounds__(256) void rowstats_kernel(
    const float* __restrict__ L, float* __restrict__ rmax,
    float* __restrict__ rinv, int NC) {
  __shared__ float red[256];
  const int r = blockIdx.x, tid = threadIdx.x;
  const float* row = L + (long)r * NC;
  float m = -INFINITY;
  for (int c = tid; c < NC; c += 256) m = fmaxf(m, row[c]);
  red[tid] = m;
  __syncthreads();
  for (int s = 128; s > 0; s >>= 1) {
    if (tid < s) red[tid] = fmaxf(red[tid], red[tid + s]);
    __syncthreads();
  }
  m = red[0];
  __syncthreads();
  float sum = 0.0f;
  for (int c = tid; c < NC; c += 256) sum += __expf(row[c] - m);
  red[tid] = sum;
  __syncthreads();
  for (int s = 128; s > 0; s >>= 1) {
    if (tid < s) red[tid] += red[tid + s];
    __syncthreads();
  }
  if (tid == 0) { rmax[r] = m; rinv[r] = 1.0f / red[0]; }
}

// Column stats: one thread per column, reads coalesced across the row.
__global__ __launch_bounds__(256) void colstats_kernel(
    const float* __restrict__ L, float* __restrict__ cmax,
    float* __restrict__ cinv, int NR, int NC) {
  const int c = blockIdx.x * 256 + threadIdx.x;
  float m = -INFINITY;
  for (int r = 0; r < NR; ++r) m = fmaxf(m, L[(long)r * NC + c]);
  float sum = 0.0f;
  for (int r = 0; r < NR; ++r) sum += __expf(L[(long)r * NC + c] - m);
  cmax[c] = m;
  cinv[c] = 1.0f / sum;
}

// ---------------------------------------------------------------------------
// O[M,N] = softmax(L) * scale @ V[K,N]
//   transposed==0: P[m][k] = exp(L[m*ldl+k] - smax[m]) * sinv[m] * s
//   transposed==1: P[m][k] = exp(L[k*ldl+m] - smax[m]) * sinv[m] * s
// Softmax rebuilt in-register during LDS staging (attn matrix never stored),
// double-buffered.
// ---------------------------------------------------------------------------
__global__ __launch_bounds__(256, 2) void attn_kernel(
    const float* __restrict__ L, const float* __restrict__ smax,
    const float* __restrict__ sinv, const float* __restrict__ scale_p,
    const __bf16* __restrict__ V, float* __restrict__ O,
    int M, int N, int K, int ldl, int transposed) {
  __shared__ __bf16 sA[2][BM * LDT];
  __shared__ __bf16 sB[2][BN * LDT];
  const int tid = threadIdx.x, lane = tid & 31, wid = tid >> 5;
  const int wm = (wid & 3) * 32, wn = (wid >> 2) * 64;
  const long bm0 = (long)blockIdx.y * BM, bn0 = (long)blockIdx.x * BN;
  const float s = scale_p[0];

  v8f acc[2][4];
#pragma unroll
  for (int i = 0; i < 2; ++i)
#pragma unroll
    for (int j = 0; j < 4; ++j)
#pragma unroll
      for (int v = 0; v < 8; ++v) acc[i][j][v] = 0.0f;

  auto stage = [&](int buf, int k0) {
    if (!transposed) {
      const int r = tid >> 1, c = (tid & 1) * 16;
      const long row = bm0 + r;
      const float* src = L + row * (long)ldl + k0 + c;
      const float mx = smax[row];
      const float sc = sinv[row] * s;
      __bf16* dst = sA[buf] + r * LDT + c;
#pragma unroll
      for (int j = 0; j < 16; ++j)
        dst[j] = (__bf16)(__expf(src[j] - mx) * sc);
    } else {
      // read L row-wise (coalesced along m), write transposed into sA
      const int kk = tid >> 3, ii = (tid & 7) * 16;
      const float* src = L + (long)(k0 + kk) * ldl + bm0 + ii;
#pragma unroll
      for (int j = 0; j < 16; ++j) {
        const long m = bm0 + ii + j;
        sA[buf][(ii + j) * LDT + kk] =
            (__bf16)(__expf(src[j] - smax[m]) * (sinv[m] * s));
      }
    }
    {  // stage B^T: sB[n][k] = V[k][n]  (coalesced global reads along n)
      const int kk = tid >> 3, nn = (tid & 7) * 16;
      const __bf16* src = V + (long)(k0 + kk) * N + bn0 + nn;
      v8bf lo = *(const v8bf*)(src);
      v8bf hi = *(const v8bf*)(src + 8);
#pragma unroll
      for (int j = 0; j < 8; ++j) {
        sB[buf][(nn + j) * LDT + kk] = lo[j];
        sB[buf][(nn + 8 + j) * LDT + kk] = hi[j];
      }
    }
  };

  stage(0, 0);
  __syncthreads();
  for (int k0 = 0; k0 < K; k0 += BK) {
    const int cur = (k0 / BK) & 1;
    if (k0 + BK < K) stage(cur ^ 1, k0 + BK);   // exp/transpose under WMMAs
    mma_step(sA[cur], sB[cur], wm, wn, lane, acc);
    __syncthreads();
  }

  const int nl = lane & 15, h = lane >> 4;
#pragma unroll
  for (int j = 0; j < 4; ++j) {
    const long col = bn0 + wn + 16 * j + nl;
#pragma unroll
    for (int i = 0; i < 2; ++i)
#pragma unroll
      for (int v = 0; v < 8; ++v) {
        const long row = bm0 + wm + 16 * i + 8 * h + v;
        O[row * (long)N + col] = acc[i][j][v];
      }
  }
}

// ---------------------------------------------------------------------------
extern "C" void kernel_launch(void* const* d_in, const int* in_sizes, int n_in,
                              void* d_out, int out_size, void* d_ws,
                              size_t ws_size, hipStream_t stream) {
  const float* text     = (const float*)d_in[0];
  const float* image    = (const float*)d_in[1];
  const float* w_text   = (const float*)d_in[2];
  const float* b_text   = (const float*)d_in[3];
  const float* w_image  = (const float*)d_in[4];
  const float* b_image  = (const float*)d_in[5];
  const float* scale_TI = (const float*)d_in[6];
  const float* scale_IT = (const float*)d_in[7];

  float* out_TI = (float*)d_out;                         // [N_T, DIM]
  float* out_IT = (float*)d_out + (long)NT_ * DIM_;      // [N_I, DIM]

  char* ws = (char*)d_ws;
  __bf16* t_bf   = (__bf16*)(ws);                        //  8 MiB
  __bf16* i_bf   = (__bf16*)(ws + (8l << 20));           //  8 MiB
  __bf16* txt_bf = (__bf16*)(ws + (16l << 20));          //  8 MiB
  __bf16* img_bf = (__bf16*)(ws + (24l << 20));          //  8 MiB
  float*  L      = (float*)(ws + (32l << 20));           // 64 MiB (L2-resident)
  float*  rmax   = (float*)(ws + (96l << 20));
  float*  rinv   = rmax + NT_;
  float*  cmax   = rinv + NT_;
  float*  cinv   = cmax + NI_;

  // bf16 copies of the raw features (V operands of the attention GEMMs)
  cvt_bf16_kernel<<<1024, 256, 0, stream>>>(text, txt_bf, (long)NT_ * DIM_);
  cvt_bf16_kernel<<<1024, 256, 0, stream>>>(image, img_bf, (long)NI_ * DIM_);

  // projections: t = text @ w_text^T + b_text ; i = image @ w_image^T + b_image
  proj_kernel<<<dim3(HID_ / BN, NT_ / BM), 256, 0, stream>>>(
      text, w_text, b_text, t_bf, NT_, HID_, DIM_);
  proj_kernel<<<dim3(HID_ / BN, NI_ / BM), 256, 0, stream>>>(
      image, w_image, b_image, i_bf, NI_, HID_, DIM_);

  // logits = (t @ i^T) / sqrt(H), kept f32 in workspace (64 MiB < 192 MiB L2)
  logits_kernel<<<dim3(NI_ / BN, NT_ / BM), 256, 0, stream>>>(
      t_bf, i_bf, L, NT_, NI_, HID_);

  // softmax statistics over both axes
  rowstats_kernel<<<NT_, 256, 0, stream>>>(L, rmax, rinv, NI_);
  colstats_kernel<<<NI_ / 256, 256, 0, stream>>>(L, cmax, cinv, NT_, NI_);

  // out_TI = softmax_rows(L)*scale_TI @ image ; out_IT = softmax_cols(L)^T*scale_IT @ text
  attn_kernel<<<dim3(DIM_ / BN, NT_ / BM), 256, 0, stream>>>(
      L, rmax, rinv, scale_TI, img_bf, out_TI, NT_, DIM_, NI_, NI_, 0);
  attn_kernel<<<dim3(DIM_ / BN, NI_ / BM), 256, 0, stream>>>(
      L, cmax, cinv, scale_IT, txt_bf, out_IT, NI_, DIM_, NT_, NI_, 1);
}
